// SlidingWindowTransformerActorCritic_88759794139215
// MI455X (gfx1250) — matrix operand order repaired and checked by reading, not compiled, not run on hardware
//
#include <hip/hip_runtime.h>
#include <hip/hip_bf16.h>

typedef _Float16 half_t;
typedef __attribute__((ext_vector_type(16))) _Float16 v16h;
typedef __attribute__((ext_vector_type(2)))  _Float16 v2h;
typedef __attribute__((ext_vector_type(8)))  float    v8f;

#define TT 256   // T
#define DD 512   // D
#define WW 32    // window
#define LL 4     // layers
#define DH 64    // head dim
#define NWIN (8*TT)

// ---- WMMA operand loaders (contiguous k = 16*(lane/16) + e in both A and B) ----

// A: 16x32 f16 tile from f16 row-major buffer (16 contiguous K per lane)
__device__ inline v16h ld_a_f16(const half_t* base, int ld, int m0, int k0, int lane) {
  int m = m0 + (lane & 15);
  const half_t* p = base + m * ld + k0 + ((lane >> 4) << 4);
  v16h a;
#pragma unroll
  for (int e = 0; e < 16; ++e) a[e] = p[e];
  return a;
}

// B: 32x16 tile where source is row-major (N,K): element (k,n) = W[n][k]
// lane reads 16 contiguous K -> 2x b128 global loads
__device__ inline v16h ld_b_nk(const half_t* Wnk, long ld, int n0, int k0, int lane) {
  const half_t* p = Wnk + (long)(n0 + (lane & 15)) * ld + k0 + ((lane >> 4) << 4);
  v16h b;
#pragma unroll
  for (int e = 0; e < 16; ++e) b[e] = p[e];
  return b;
}

// B: 32x16 tile where source is row-major (K,N) with row stride ld
__device__ inline v16h ld_b_kn(const half_t* base, int ld, int n0, int k0, int lane) {
  int col = n0 + (lane & 15);
  int kb  = k0 + ((lane >> 4) << 4);
  v16h b;
#pragma unroll
  for (int e = 0; e < 16; ++e) b[e] = base[(kb + e) * ld + col];
  return b;
}

__device__ inline v8f wmma_step(v16h a, v16h b, v8f c) {
  return __builtin_amdgcn_wmma_f32_16x16x32_f16(false, a, false, b, (short)0, c, false, false);
}

__device__ inline float wave_sum(float v) {
#pragma unroll
  for (int off = 16; off; off >>= 1) v += __shfl_xor(v, off, 32);
  return v;
}

// fp32 -> fp16 weight conversion (one-time per launch, weights stay L2-resident)
__global__ void cvt_f32_f16_kernel(const float* __restrict__ src, half_t* __restrict__ dst, long n) {
  long i = (long)blockIdx.x * blockDim.x + threadIdx.x;
  long stride = (long)gridDim.x * blockDim.x;
  for (; i < n; i += stride) dst[i] = (half_t)src[i];
}

__global__ __launch_bounds__(256)
void swt_forward_kernel(const float* __restrict__ feats, const float* __restrict__ pos,
                        const half_t* __restrict__ WqkvH, const float* __restrict__ bqkv,
                        const half_t* __restrict__ WoH,   const float* __restrict__ bo,
                        const float* __restrict__ ln1g,   const float* __restrict__ ln1b,
                        const half_t* __restrict__ W1H,   const float* __restrict__ b1,
                        const half_t* __restrict__ W2H,   const float* __restrict__ b2,
                        const float* __restrict__ ln2g,   const float* __restrict__ ln2b,
                        const float* __restrict__ Wp,     const float* __restrict__ bp,
                        const float* __restrict__ Wv,     const float* __restrict__ bv,
                        float* __restrict__ out) {
  __shared__ float  X[WW * DD];            // 64 KB  activation (fp32 residual stream)
  __shared__ half_t XH[WW * DD];           // 32 KB  f16 mirror of X (WMMA A operand)
  __shared__ half_t QKV[WW * 3 * DD];      // 96 KB  qkv (f16); reused as FFN hidden chunk
  __shared__ float  Yb[WW * DD];           // 64 KB  matmul out (fp32); low half = scores,
                                           //        high half reused as f16 attn probs
  __shared__ half_t CTX[WW * DD];          // 32 KB  attention context (f16)

  const int nwin = blockIdx.x;             // window id = b*T + t
  const int bIdx = nwin / TT;
  const int tw   = nwin % TT;
  const int tid  = threadIdx.x;
  const int lane = tid & 31;
  const int wave = tid >> 5;               // 8 waves

  half_t* PH = (half_t*)(Yb + 8 * WW * WW);  // f16 probs region (upper 32KB of Yb)

  // gather sliding window + pos embed (f32 + f16 mirror)
  for (int i = tid; i < WW * DD; i += 256) {
    int w = i >> 9;                        // / DD
    int d = i & (DD - 1);
    int src = tw + 1 - WW + w; if (src < 0) src = 0;
    float v = feats[((long)bIdx * TT + src) * DD + d] + pos[i];
    X[i]  = v;
    XH[i] = (half_t)v;
  }
  __syncthreads();

  for (int l = 0; l < LL; ++l) {
    // ---------------- QKV projection: (32x512)@(512x1536) ----------------
    {
      const half_t* Wq = WqkvH + (long)l * 3 * DD * DD;
      const float*  bq = bqkv + l * 3 * DD;
      for (int t = wave; t < 2 * 96; t += 8) {
        int m0 = (t & 1) << 4, n0 = (t >> 1) << 4;
        v8f acc = {};
        for (int k0 = 0; k0 < DD; k0 += 32) {
          v16h a  = ld_a_f16(XH, DD, m0, k0, lane);
          v16h bb = ld_b_nk(Wq, DD, n0, k0, lane);
          acc = wmma_step(a, bb, acc);
        }
        int nn = n0 + (lane & 15), hf = lane >> 4;
        float bias = bq[nn];
#pragma unroll
        for (int r = 0; r < 8; ++r)
          QKV[(m0 + r + 8 * hf) * (3 * DD) + nn] = (half_t)(acc[r] + bias);
      }
    }
    __syncthreads();

    // ---------------- attention scores: one head per wave ----------------
    {
      const half_t* qh = QKV + wave * DH;
      const half_t* kh = QKV + DD + wave * DH;
      float* S = Yb + wave * (WW * WW);
#pragma unroll
      for (int t = 0; t < 4; ++t) {
        int m0 = (t & 1) << 4, n0 = (t >> 1) << 4;
        v8f acc = {};
#pragma unroll
        for (int k0 = 0; k0 < DH; k0 += 32) {
          v16h a  = ld_a_f16(qh, 3 * DD, m0, k0, lane);
          v16h bb = ld_b_nk(kh, (long)3 * DD, n0, k0, lane);
          acc = wmma_step(a, bb, acc);
        }
        int nn = n0 + (lane & 15), hf = lane >> 4;
#pragma unroll
        for (int r = 0; r < 8; ++r)
          S[(m0 + r + 8 * hf) * WW + nn] = acc[r];
      }
    }
    __syncthreads();

    // causal softmax (scale 1/sqrt(64)); 256 threads = 8 heads x 32 rows
    // reads f32 scores, writes f16 probs (contiguous per-row packed stores)
    {
      float*  S = Yb + wave * (WW * WW) + lane * WW;
      half_t* P = PH + wave * (WW * WW) + lane * WW;
      int m = lane;
      float mx = -3.4e38f;
      for (int j = 0; j <= m; ++j) { float s = S[j] * 0.125f; S[j] = s; mx = fmaxf(mx, s); }
      float sum = 0.f;
      for (int j = 0; j <= m; ++j) { float e = __expf(S[j] - mx); S[j] = e; sum += e; }
      float inv = 1.f / sum;
      for (int j = 0; j <= m; ++j) P[j] = (half_t)(S[j] * inv);
      for (int j = m + 1; j < WW; ++j) P[j] = (half_t)0.f;
    }
    __syncthreads();

    // ---------------- ctx = attn @ v (per head, K=32 = single step) ----------------
    {
      const half_t* P  = PH + wave * (WW * WW);
      const half_t* vh = QKV + 2 * DD + wave * DH;
#pragma unroll
      for (int t = 0; t < 8; ++t) {
        int m0 = (t & 1) << 4, n0 = (t >> 1) << 4;      // n0: 0..48
        v16h a  = ld_a_f16(P, WW, m0, 0, lane);
        v16h bb = ld_b_kn(vh, 3 * DD, n0, 0, lane);
        v8f acc = {};
        acc = wmma_step(a, bb, acc);
        int nn = wave * DH + n0 + (lane & 15), hf = lane >> 4;
#pragma unroll
        for (int r = 0; r < 8; ++r)
          CTX[(m0 + r + 8 * hf) * DD + nn] = (half_t)acc[r];
      }
    }
    __syncthreads();

    // ---------------- out projection: (32x512)@(512x512) ----------------
    {
      const half_t* Wol = WoH + (long)l * DD * DD;
      const float*  bol = bo + l * DD;
      for (int t = wave; t < 64; t += 8) {
        int m0 = (t & 1) << 4, n0 = (t >> 1) << 4;
        v8f acc = {};
        for (int k0 = 0; k0 < DD; k0 += 32) {
          v16h a  = ld_a_f16(CTX, DD, m0, k0, lane);
          v16h bb = ld_b_nk(Wol, DD, n0, k0, lane);
          acc = wmma_step(a, bb, acc);
        }
        int nn = n0 + (lane & 15), hf = lane >> 4;
        float bias = bol[nn];
#pragma unroll
        for (int r = 0; r < 8; ++r)
          Yb[(m0 + r + 8 * hf) * DD + nn] = acc[r] + bias;
      }
    }
    __syncthreads();

    // residual + LayerNorm1 -> X, XH  (4 rows per wave, paired lanes for packed f16 store)
    {
      const float* g  = ln1g + l * DD;
      const float* bt = ln1b + l * DD;
      for (int r = wave * 4; r < wave * 4 + 4; ++r) {
        float* xr = X + r * DD; const float* yr = Yb + r * DD;
        float s = 0.f, s2 = 0.f;
        for (int d = lane * 2; d < DD; d += 64) {
          float v0 = xr[d] + yr[d], v1 = xr[d + 1] + yr[d + 1];
          s += v0 + v1; s2 += v0 * v0 + v1 * v1;
        }
        s = wave_sum(s); s2 = wave_sum(s2);
        float mu  = s * (1.f / DD);
        float inv = rsqrtf(s2 * (1.f / DD) - mu * mu + 1e-5f);
        for (int d = lane * 2; d < DD; d += 64) {
          float v0 = xr[d] + yr[d], v1 = xr[d + 1] + yr[d + 1];
          float o0 = (v0 - mu) * inv * g[d] + bt[d];
          float o1 = (v1 - mu) * inv * g[d + 1] + bt[d + 1];
          xr[d] = o0; xr[d + 1] = o1;
          *(v2h*)(XH + r * DD + d) = (v2h){(_Float16)o0, (_Float16)o1};
        }
      }
    }
    __syncthreads();

    // ---------------- FFN: 512 -> 2048 (relu) -> 512, hidden chunked by 512 ----------------
    {
      half_t* HB = QKV;                     // reuse as 32x512 f16 hidden chunk
      v8f zero = {};
      v8f facc[8];
#pragma unroll
      for (int j = 0; j < 8; ++j) facc[j] = zero;

      for (int c = 0; c < 4; ++c) {
        const half_t* W1l = W1H + (long)l * 4 * DD * DD + (long)c * DD * DD;
        const float*  b1l = b1 + l * 4 * DD + c * DD;
        for (int t = wave; t < 64; t += 8) {
          int m0 = (t & 1) << 4, n0 = (t >> 1) << 4;
          v8f acc = {};
          for (int k0 = 0; k0 < DD; k0 += 32) {
            v16h a  = ld_a_f16(XH, DD, m0, k0, lane);
            v16h bb = ld_b_nk(W1l, DD, n0, k0, lane);
            acc = wmma_step(a, bb, acc);
          }
          int nn = n0 + (lane & 15), hf = lane >> 4;
          float bias = b1l[nn];
#pragma unroll
          for (int r = 0; r < 8; ++r)
            HB[(m0 + r + 8 * hf) * DD + nn] = (half_t)fmaxf(acc[r] + bias, 0.f);
        }
        __syncthreads();

        // accumulate this K-chunk into register-resident output tiles
        const half_t* W2l = W2H + (long)l * DD * 4 * DD + c * DD;
#pragma unroll
        for (int j = 0; j < 8; ++j) {
          int t = wave * 8 + j;
          int m0 = (t & 1) << 4, n0 = (t >> 1) << 4;
          for (int k0 = 0; k0 < DD; k0 += 32) {
            v16h a  = ld_a_f16(HB, DD, m0, k0, lane);
            v16h bb = ld_b_nk(W2l, (long)4 * DD, n0, k0, lane);
            facc[j] = wmma_step(a, bb, facc[j]);
          }
        }
        __syncthreads();
      }

      const float* b2l = b2 + l * DD;
#pragma unroll
      for (int j = 0; j < 8; ++j) {
        int t = wave * 8 + j;
        int m0 = (t & 1) << 4, n0 = (t >> 1) << 4;
        int nn = n0 + (lane & 15), hf = lane >> 4;
        float bias = b2l[nn];
#pragma unroll
        for (int r = 0; r < 8; ++r)
          Yb[(m0 + r + 8 * hf) * DD + nn] = facc[j][r] + bias;
      }
    }
    __syncthreads();

    // residual + LayerNorm2 -> X, XH
    {
      const float* g  = ln2g + l * DD;
      const float* bt = ln2b + l * DD;
      for (int r = wave * 4; r < wave * 4 + 4; ++r) {
        float* xr = X + r * DD; const float* yr = Yb + r * DD;
        float s = 0.f, s2 = 0.f;
        for (int d = lane * 2; d < DD; d += 64) {
          float v0 = xr[d] + yr[d], v1 = xr[d + 1] + yr[d + 1];
          s += v0 + v1; s2 += v0 * v0 + v1 * v1;
        }
        s = wave_sum(s); s2 = wave_sum(s2);
        float mu  = s * (1.f / DD);
        float inv = rsqrtf(s2 * (1.f / DD) - mu * mu + 1e-5f);
        for (int d = lane * 2; d < DD; d += 64) {
          float v0 = xr[d] + yr[d], v1 = xr[d + 1] + yr[d + 1];
          float o0 = (v0 - mu) * inv * g[d] + bt[d];
          float o1 = (v1 - mu) * inv * g[d + 1] + bt[d + 1];
          xr[d] = o0; xr[d + 1] = o1;
          *(v2h*)(XH + r * DD + d) = (v2h){(_Float16)o0, (_Float16)o1};
        }
      }
    }
    __syncthreads();
  }

  // ---------------- actor/critic head on last window row ----------------
  if (wave == 0) {
    const float* xr = X + (WW - 1) * DD;
    for (int o = 0; o < 7; ++o) {
      const float* wr = (o < 6) ? (Wp + o * DD) : Wv;
      float s = 0.f;
      for (int d = lane; d < DD; d += 32) s += xr[d] * wr[d];
      s = wave_sum(s);
      if (lane == 0) {
        if (o < 6) out[(long)nwin * 6 + o] = s + bp[o];
        else       out[(long)NWIN * 6 + nwin] = s + bv[0];
      }
    }
  }
}

extern "C" void kernel_launch(void* const* d_in, const int* in_sizes, int n_in,
                              void* d_out, int out_size, void* d_ws, size_t ws_size,
                              hipStream_t stream) {
  (void)in_sizes; (void)n_in; (void)out_size; (void)ws_size;
  const float* feats = (const float*)d_in[0];
  const float* pos   = (const float*)d_in[1];
  const float* Wqkv  = (const float*)d_in[2];
  const float* bqkv  = (const float*)d_in[3];
  const float* Wo    = (const float*)d_in[4];
  const float* bo    = (const float*)d_in[5];
  const float* ln1g  = (const float*)d_in[6];
  const float* ln1b  = (const float*)d_in[7];
  const float* W1    = (const float*)d_in[8];
  const float* b1    = (const float*)d_in[9];
  const float* W2    = (const float*)d_in[10];
  const float* b2    = (const float*)d_in[11];
  const float* ln2g  = (const float*)d_in[12];
  const float* ln2b  = (const float*)d_in[13];
  const float* Wp    = (const float*)d_in[14];
  const float* bp    = (const float*)d_in[15];
  const float* Wv    = (const float*)d_in[16];
  const float* bv    = (const float*)d_in[17];

  // workspace: f16 weight mirrors (~25 MB), L2-resident across the whole launch
  const long nQ = (long)LL * 3 * DD * DD;   // 3,145,728
  const long nO = (long)LL * DD * DD;       // 1,048,576
  const long n1 = (long)LL * 4 * DD * DD;   // 4,194,304
  const long n2 = n1;
  half_t* WqkvH = (half_t*)d_ws;
  half_t* WoH   = WqkvH + nQ;
  half_t* W1H   = WoH + nO;
  half_t* W2H   = W1H + n1;

  cvt_f32_f16_kernel<<<512, 256, 0, stream>>>(Wqkv, WqkvH, nQ);
  cvt_f32_f16_kernel<<<512, 256, 0, stream>>>(Wo,   WoH,   nO);
  cvt_f32_f16_kernel<<<512, 256, 0, stream>>>(W1,   W1H,   n1);
  cvt_f32_f16_kernel<<<512, 256, 0, stream>>>(W2,   W2H,   n2);

  swt_forward_kernel<<<NWIN, 256, 0, stream>>>(
      feats, pos, WqkvH, bqkv, WoH, bo, ln1g, ln1b,
      W1H, b1, W2H, b2, ln2g, ln2b, Wp, bp, Wv, bv, (float*)d_out);
}
